// Int8Linear_91147795955912
// MI455X (gfx1250) — matrix operand order repaired and checked by reading, not compiled, not run on hardware
//
#include <hip/hip_runtime.h>
#include <hip/hip_bf16.h>

typedef __attribute__((ext_vector_type(16))) _Float16 v16h;
typedef __attribute__((ext_vector_type(8)))  _Float16 v8h;
typedef __attribute__((ext_vector_type(4)))  _Float16 v4h;
typedef __attribute__((ext_vector_type(8)))  float    v8f;

#define TILE_M  128
#define TILE_N  128
#define TILE_K  32
#define LDSROW  40          // TILE_K + 8 halves padding (80B row stride, 16B aligned)
#define THREADS 256

union V16U { v16h v; v8h h[2]; };

// ---------------------------------------------------------------------------
// CDNA5 async global->LDS DMA (ASYNCcnt path): per-lane 16B copy, no VGPR
// staging, no ds_store in the WMMA issue stream.
//   vdst  = VGPR with LDS byte address (low 32 bits of generic shared ptr)
//   vaddr = VGPR pair with 64-bit global address
// ---------------------------------------------------------------------------
__device__ __forceinline__ void async_copy_b128(const _Float16* gsrc,
                                                const _Float16* ldst) {
  unsigned lds_off = (unsigned)(size_t)ldst;   // aperture: addr[31:0] = LDS offset
  asm volatile("global_load_async_to_lds_b128 %0, %1, off"
               :: "v"(lds_off), "v"(gsrc)
               : "memory");
}

__device__ __forceinline__ void wait_async0() {
#if __has_builtin(__builtin_amdgcn_s_wait_asynccnt)
  __builtin_amdgcn_s_wait_asynccnt(0);
#else
  asm volatile("s_wait_asynccnt 0x0" ::: "memory");
#endif
}

// ---------------------------------------------------------------------------
// Conversion kernels: one-time pass so the GEMM streams f16 (157 MB total,
// L2-resident on the 192MB L2) instead of f32/int32 (314 MB).
// ---------------------------------------------------------------------------
__global__ __launch_bounds__(256) void cvt_x_f32_to_f16(
    const float* __restrict__ in, _Float16* __restrict__ out, long long n4) {
  long long i = (long long)blockIdx.x * blockDim.x + threadIdx.x;
  if (i >= n4) return;
  float4 v = ((const float4*)in)[i];
  v4h o;
  o.x = (_Float16)v.x; o.y = (_Float16)v.y;
  o.z = (_Float16)v.z; o.w = (_Float16)v.w;
  ((v4h*)out)[i] = o;
}

__global__ __launch_bounds__(256) void cvt_w_i32_to_f16(
    const int* __restrict__ in, _Float16* __restrict__ out, long long n4) {
  long long i = (long long)blockIdx.x * blockDim.x + threadIdx.x;
  if (i >= n4) return;
  int4 v = ((const int4*)in)[i];
  v4h o;                               // int8 range is exact in f16
  o.x = (_Float16)(float)v.x; o.y = (_Float16)(float)v.y;
  o.z = (_Float16)(float)v.z; o.w = (_Float16)(float)v.w;
  ((v4h*)out)[i] = o;
}

// ---------------------------------------------------------------------------
// WMMA f16 GEMM with per-output-channel scale epilogue.
//   A  : [M,K] f16 row-major (activations)
//   W  : [N,K] f16 row-major (int-valued weights) -> serves as K x N B-matrix
//   C  : [M,N] f32,  C[m,n] = scale[n] * sum_k A[m,k]*W[n,k]
// ---------------------------------------------------------------------------
__global__ __launch_bounds__(THREADS) void int8_linear_wmma(
    const _Float16* __restrict__ A,
    const _Float16* __restrict__ W,
    const float*    __restrict__ scale,
    float*          __restrict__ C,
    int M, int N, int K)
{
  __shared__ _Float16 lds[2 * 2 * TILE_M * LDSROW];   // [buf][A|B][row][LDSROW]
  const int bufStride = 2 * TILE_M * LDSROW;

  const int tid  = threadIdx.x;
  const int lane = tid & 31;
  const int wave = tid >> 5;
  const int wm   = (wave & 1) * 64;    // wave M offset inside block tile
  const int wn   = (wave >> 1) * 32;   // wave N offset inside block tile
  const int l15  = lane & 15;
  const int hi   = lane >> 4;          // lane group 0/1

  const int bn = blockIdx.x * TILE_N;
  const int bm = blockIdx.y * TILE_M;

  // global->LDS copy mapping: 512 16B-chunks per matrix tile, 2 per thread
  const int lrow = tid >> 2;           // 0..63 (and +64 for second chunk)
  const int lcol = (tid & 3) * 8;      // half offset 0,8,16,24

  const _Float16* Arow0 = A + (size_t)(bm + lrow) * K + lcol;
  const _Float16* Arow1 = Arow0 + (size_t)64 * K;
  const _Float16* Wrow0 = W + (size_t)(bn + lrow) * K + lcol;
  const _Float16* Wrow1 = Wrow0 + (size_t)64 * K;

  // ---- preload tile 0 into buffer 0 via async DMA
  {
    _Float16* Ab = lds;
    _Float16* Bb = lds + TILE_M * LDSROW;
    async_copy_b128(Arow0, Ab + lrow * LDSROW + lcol);
    async_copy_b128(Arow1, Ab + (lrow + 64) * LDSROW + lcol);
    async_copy_b128(Wrow0, Bb + lrow * LDSROW + lcol);
    async_copy_b128(Wrow1, Bb + (lrow + 64) * LDSROW + lcol);
  }
  wait_async0();
  __syncthreads();

  v8f acc[4][2];
#pragma unroll
  for (int i = 0; i < 4; ++i) { acc[i][0] = {}; acc[i][1] = {}; }

  const int NK = K / TILE_K;
  for (int kc = 0; kc < NK; ++kc) {
    const int cur = kc & 1;
    const _Float16* Ab = lds + cur * bufStride;
    const _Float16* Bb = Ab + TILE_M * LDSROW;

    // kick off next tile's DMA into the other buffer (safe: that buffer was
    // last read in iteration kc-1, fenced by its end-of-iteration barrier)
    const bool more = (kc + 1) < NK;
    if (more) {
      const int kb = (kc + 1) * TILE_K;
      _Float16* An = lds + (cur ^ 1) * bufStride;
      _Float16* Bn = An + TILE_M * LDSROW;
      async_copy_b128(Arow0 + kb, An + lrow * LDSROW + lcol);
      async_copy_b128(Arow1 + kb, An + (lrow + 64) * LDSROW + lcol);
      async_copy_b128(Wrow0 + kb, Bn + lrow * LDSROW + lcol);
      async_copy_b128(Wrow1 + kb, Bn + (lrow + 64) * LDSROW + lcol);
    }

    // B fragments (32x16 KxN layout): 16 contiguous K-halves, column n = l15
    V16U bf0, bf1;
    {
      const _Float16* bp0 = Bb + (wn + l15) * LDSROW + hi * 16;
      bf0.h[0] = *(const v8h*)(bp0);
      bf0.h[1] = *(const v8h*)(bp0 + 8);
      const _Float16* bp1 = bp0 + 16 * LDSROW;
      bf1.h[0] = *(const v8h*)(bp1);
      bf1.h[1] = *(const v8h*)(bp1 + 8);
    }

#pragma unroll
    for (int i = 0; i < 4; ++i) {
      // A fragment (16x32 MxK layout): row m = l15, K lo/hi split by lane group
      V16U af;
      const _Float16* ap = Ab + (wm + i * 16 + l15) * LDSROW;
      af.h[0] = *(const v8h*)(ap + hi * 16);       // K = 16*hi + 0..7
      af.h[1] = *(const v8h*)(ap + hi * 16 + 8);   // K = 16*hi + 8..15
      acc[i][0] = __builtin_amdgcn_wmma_f32_16x16x32_f16(
          false, af.v, false, bf0.v, (short)0, acc[i][0], false, false);
      acc[i][1] = __builtin_amdgcn_wmma_f32_16x16x32_f16(
          false, af.v, false, bf1.v, (short)0, acc[i][1], false, false);
    }

    // own async DMAs must be LDS-visible before the barrier publishes buffer
    if (more) wait_async0();
    __syncthreads();
  }

  // ---- epilogue: per-channel dequant scale, f32 store
#pragma unroll
  for (int j = 0; j < 2; ++j) {
    const int n = bn + wn + j * 16 + l15;
    const float s = scale[n];
#pragma unroll
    for (int i = 0; i < 4; ++i) {
      const int mbase = bm + wm + i * 16 + hi * 8;   // C/D layout: m = vgpr + 8*hi
      float* cp = C + (size_t)mbase * N + n;
#pragma unroll
      for (int v = 0; v < 8; ++v) {
        cp[(size_t)v * N] = acc[i][j][v] * s;
      }
    }
  }
}

// ---------------------------------------------------------------------------
extern "C" void kernel_launch(void* const* d_in, const int* in_sizes, int n_in,
                              void* d_out, int out_size, void* d_ws, size_t ws_size,
                              hipStream_t stream) {
  const float* x     = (const float*)d_in[0];
  const int*   w     = (const int*)d_in[1];
  const float* scale = (const float*)d_in[2];
  float*       out   = (float*)d_out;

  const long long xcount = in_sizes[0];          // M*K
  const long long wcount = in_sizes[1];          // N*K
  const int N = in_sizes[2];
  const int K = (int)(wcount / N);
  const int M = (int)(xcount / K);

  _Float16* xh = (_Float16*)d_ws;                // 2*M*K bytes
  _Float16* wh = xh + xcount;                    // 2*N*K bytes (16B aligned)

  const long long x4 = xcount / 4;
  const long long w4 = wcount / 4;
  cvt_x_f32_to_f16<<<(unsigned)((x4 + 255) / 256), 256, 0, stream>>>(x, xh, x4);
  cvt_w_i32_to_f16<<<(unsigned)((w4 + 255) / 256), 256, 0, stream>>>(w, wh, w4);

  dim3 grid(N / TILE_N, M / TILE_M);             // 86 x 64, exact
  int8_linear_wmma<<<grid, THREADS, 0, stream>>>(xh, wh, scale, out, M, N, K);
}
